// xMultiheadAttention_48911087567063
// MI455X (gfx1250) — matrix operand order repaired and checked by reading, not compiled, not run on hardware
//
#include <hip/hip_runtime.h>
#include <hip/hip_bf16.h>

typedef __attribute__((ext_vector_type(16))) _Float16 v16h;
typedef __attribute__((ext_vector_type(8)))  _Float16 v8h;
typedef __attribute__((ext_vector_type(8)))  float    v8f;
typedef __attribute__((ext_vector_type(4)))  unsigned int u32x4;
typedef __attribute__((ext_vector_type(8)))  int      i32x8;
typedef __attribute__((ext_vector_type(4)))  int      i32x4;

// ---------- constants (fixed problem size) ----------
#define TT 1024      // sequence length
#define BB 8         // batch
#define DD 1024      // model dim
#define HH 16        // heads
#define HD 64        // head dim
#define MROWS (TT*BB)
#define NEG_LOG_BASE_OVER_HALF 0.28782313662425574f  // ln(10000)/32

#if defined(__has_builtin)
#  if __has_builtin(__builtin_amdgcn_tensor_load_to_lds)
#    define HAS_TDM 1
#  endif
#endif
#ifndef HAS_TDM
#  define HAS_TDM 0
#endif

// ---------- WMMA helpers ----------
static __device__ inline v8f wmma16(v16h a, v16h b, v8f c) {
  return __builtin_amdgcn_wmma_f32_16x16x32_f16(
      false, a, false, b, (short)0, c, false, false);
}

// A fragment 16x32 (MxK), row-major [m][k].
static __device__ inline v16h load_a16x32(const _Float16* __restrict__ base,
                                          int ld, int lane) {
  int m  = lane & 15;
  int kb = (lane >> 4) << 3;
  union { v16h v; v8h h[2]; } u;
  const _Float16* p = base + (size_t)m * ld + kb;
  u.h[0] = *(const v8h*)(p);
  u.h[1] = *(const v8h*)(p + 16);
  return u.v;
}

// B fragment 32x16 (KxN) from row-major [n][k] storage.
static __device__ inline v16h load_b32x16(const _Float16* __restrict__ base,
                                          int ld, int lane) {
  int n  = lane & 15;
  int kg = (lane >> 4) << 3;
  union { v16h v; v8h h[2]; } u;
  const _Float16* p = base + (size_t)n * ld + kg;
  u.h[0] = *(const v8h*)(p);
  u.h[1] = *(const v8h*)(p + 16);
  return u.v;
}

// B fragment from LDS-staged tile (row pitch 128 halves).
static __device__ inline v16h load_b_lds(const _Float16* tile, int row0,
                                         int koff, int lane) {
  int n  = lane & 15;
  int kg = (lane >> 4) << 3;
  const _Float16* p = tile + (size_t)(row0 + n) * 128 + koff + kg;
  union { v16h v; v8h h[2]; } u;
  u.h[0] = *(const v8h*)(p);
  u.h[1] = *(const v8h*)(p + 16);
  return u.v;
}

// ---------- TDM staging: 64 rows x 128 halves, row stride src_ld ----------
static __device__ inline void stage_b_tile(_Float16* ldst,
                                           const _Float16* src,
                                           int tid, int src_ld) {
#if HAS_TDM
  if (tid < 32) {   // one wave issues the DMA
    unsigned loff = (unsigned)(unsigned long long)(uintptr_t)ldst;
    unsigned long long ga = (unsigned long long)(uintptr_t)src;
    u32x4 g0;
    g0[0] = 1u;                                   // count=1
    g0[1] = loff;                                 // lds_addr
    g0[2] = (unsigned)(ga & 0xffffffffu);
    g0[3] = (unsigned)((ga >> 32) & 0x01ffffffu) | 0x80000000u;  // type=2
    i32x8 g1;
    g1[0] = (int)(1u << 16);                      // data_size = 2B
    g1[1] = (int)(((unsigned)src_ld & 0xffffu) << 16);
    g1[2] = (int)((((unsigned)src_ld >> 16) & 0xffffu) | 0xffff0000u);
    g1[3] = (int)(128u << 16);                    // tile_dim0 = 128 halves
    g1[4] = (int)64;                              // tile_dim1 = 64 rows
    g1[5] = (int)(unsigned)src_ld;                // tensor_dim0_stride
    g1[6] = 0;
    g1[7] = 0;
    i32x4 g2 = {0, 0, 0, 0};
    i32x4 g3 = {0, 0, 0, 0};
#if defined(__clang_major__) && (__clang_major__ >= 23)
    i32x8 g4 = {0, 0, 0, 0, 0, 0, 0, 0};
    __builtin_amdgcn_tensor_load_to_lds(g0, g1, g2, g3, g4, 0);
#else
    __builtin_amdgcn_tensor_load_to_lds(g0, g1, g2, g3, 0);
#endif
  }
#else
  for (int i = tid; i < (64 * 128) / 8; i += 256) {
    int e = i * 8, row = e >> 7, col = e & 127;
    *(v8h*)(ldst + e) = *(const v8h*)(src + (size_t)row * src_ld + col);
  }
#endif
}

static __device__ inline void stage_fence() {
#if HAS_TDM
  __builtin_amdgcn_s_wait_tensorcnt(0);
#endif
}

// ---------- fp32 -> f16 conversion ----------
__global__ void cvt_f32_to_f16(const float* __restrict__ s,
                               _Float16* __restrict__ d, int n) {
  int i = blockIdx.x * blockDim.x + threadIdx.x;
  if (i < n) d[i] = (_Float16)s[i];
}

// ==========================================================================
// Shared GEMM core: per wave 32 rows (2 m-tiles) x 64 cols (4 n-tiles),
// B tile TDM-staged through double-buffered LDS, software-pipelined frags.
// Accumulators returned in c[2][4].
// ==========================================================================
static __device__ inline void gemm_core(const _Float16* __restrict__ Arows, // wave's A base [32 x DD]
                                        const _Float16* __restrict__ wbase, // block's W rows [64 x DD]
                                        _Float16* bsm,                      // [2][64*128] LDS
                                        int tid, int lane, v8f c[2][4]) {
  stage_b_tile(bsm, wbase, tid, DD);
  stage_fence();
  __syncthreads();
  for (int kc = 0; kc < DD; kc += 128) {
    const int cur = (kc >> 7) & 1;
    _Float16* tile = bsm + cur * (64 * 128);
    if (kc + 128 < DD)
      stage_b_tile(bsm + (cur ^ 1) * (64 * 128), wbase + kc + 128, tid, DD);
    // --- software-pipelined fragment loads over 4 K-substeps ---
    v16h a0[2], b0[4];
    a0[0] = load_a16x32(Arows + kc, DD, lane);
    a0[1] = load_a16x32(Arows + (size_t)16 * DD + kc, DD, lane);
#pragma unroll
    for (int j = 0; j < 4; ++j) b0[j] = load_b_lds(tile, j * 16, 0, lane);
#pragma unroll
    for (int s = 0; s < 4; ++s) {
      v16h a1[2], b1[4];
      if (s < 3) {
        a1[0] = load_a16x32(Arows + kc + (s + 1) * 32, DD, lane);
        a1[1] = load_a16x32(Arows + (size_t)16 * DD + kc + (s + 1) * 32, DD, lane);
#pragma unroll
        for (int j = 0; j < 4; ++j)
          b1[j] = load_b_lds(tile, j * 16, (s + 1) * 32, lane);
      }
#pragma unroll
      for (int j = 0; j < 4; ++j) {
        c[0][j] = wmma16(a0[0], b0[j], c[0][j]);
        c[1][j] = wmma16(a0[1], b0[j], c[1][j]);
      }
      if (s < 3) {
        a0[0] = a1[0]; a0[1] = a1[1];
#pragma unroll
        for (int j = 0; j < 4; ++j) b0[j] = b1[j];
      }
    }
    stage_fence();
    __syncthreads();
  }
}

// ---------- QKV projection GEMM + bias + RoPE + head reshape ----------
// grid: (MROWS/256, DD/64, 3)  block: 256 (8 waves, 32 rows each)
__global__ __launch_bounds__(256) void qkv_rope_gemm(
    const _Float16* __restrict__ X,    // [3][MROWS, DD]
    const _Float16* __restrict__ W,    // [3*DD, DD]
    const float*    __restrict__ bias, // [3*DD]
    _Float16* __restrict__ Qd, _Float16* __restrict__ Kd,
    _Float16* __restrict__ Vt) {
  __shared__ __align__(16) _Float16 bsm[2 * 64 * 128];
  const int tid   = threadIdx.x;
  const int z     = blockIdx.z;
  const int lane  = tid & 31;
  const int wave  = tid >> 5;
  const int mbase = blockIdx.x * 256 + wave * 32;
  const int nbase = blockIdx.y * 64;
  const int h     = blockIdx.y;
  const _Float16* Arows = X + (size_t)z * MROWS * DD + (size_t)mbase * DD;
  const _Float16* wbase = W + (size_t)(z * DD + nbase) * DD;

  v8f c[2][4] = {};
  gemm_core(Arows, wbase, bsm, tid, lane, c);

  const int half = lane >> 4;
  float bv[4];
#pragma unroll
  for (int j = 0; j < 4; ++j) bv[j] = bias[z * DD + nbase + j * 16 + (lane & 15)];
#pragma unroll
  for (int mt = 0; mt < 2; ++mt)
#pragma unroll
    for (int j = 0; j < 4; ++j)
#pragma unroll
      for (int r = 0; r < 8; ++r) c[mt][j][r] += bv[j];

  if (z == 2) {
#pragma unroll
    for (int mt = 0; mt < 2; ++mt)
#pragma unroll
      for (int j = 0; j < 4; ++j)
#pragma unroll
        for (int r = 0; r < 8; ++r) {
          int m = mbase + mt * 16 + r + 8 * half;   // m = t*BB + b
          int t = m >> 3, b = m & 7;
          int i = j * 16 + (lane & 15);
          Vt[(((size_t)(b * HH + h)) * HD + i) * TT + t] = (_Float16)c[mt][j][r];
        }
  } else {
    _Float16* dst = (z == 0) ? Qd : Kd;
#pragma unroll
    for (int mt = 0; mt < 2; ++mt)
#pragma unroll
      for (int j = 0; j < 4; ++j)
#pragma unroll
        for (int r = 0; r < 8; ++r) {
          int i = j * 16 + (lane & 15);
          float partner = c[mt][j ^ 2][r];          // rotate-half partner
          float rot = (i < 32) ? -partner : partner;
          int m = mbase + mt * 16 + r + 8 * half;
          int t = m >> 3, b = m & 7;
          float invf = __expf(-(float)(i & 31) * NEG_LOG_BASE_OVER_HALF);
          float ang  = (float)t * invf;
          float val  = c[mt][j][r] * __cosf(ang) + rot * __sinf(ang);
          dst[(((size_t)(b * HH + h)) * TT + t) * HD + i] = (_Float16)val;
        }
  }
}

// ---------- flash attention: per wave 16 queries x all 1024 keys ----------
__global__ __launch_bounds__(256) void attn_kernel(
    const _Float16* __restrict__ Q,   // [b,h,s,hd]
    const _Float16* __restrict__ K,   // [b,h,s,hd]
    const _Float16* __restrict__ Vt,  // [b,h,hd,s]
    _Float16* __restrict__ Oa) {      // [b,s,D]
  __shared__ __align__(16) _Float16 lds[8 * 16 * HD];
  const int lane = threadIdx.x & 31;
  const int wave = threadIdx.x >> 5;
  const int bh = blockIdx.y, b = bh >> 4, h = bh & 15;
  const int qbase = blockIdx.x * 128 + wave * 16;
  const _Float16* Qp = Q + ((size_t)bh * TT + qbase) * HD;
  const _Float16* Kp = K + (size_t)bh * TT * HD;
  const _Float16* Vp = Vt + (size_t)bh * HD * TT;
  _Float16* myl = lds + wave * 16 * HD;
  const int half = lane >> 4;

  v16h qa0 = load_a16x32(Qp + 0, HD, lane);
  v16h qa1 = load_a16x32(Qp + 32, HD, lane);

  v8f o[4] = {};
  float mrow[8], lrow[8];
#pragma unroll
  for (int r = 0; r < 8; ++r) { mrow[r] = -1e30f; lrow[r] = 0.f; }

  for (int kk = 0; kk < TT; kk += 64) {
    // ---- S = Q K^T, depth-1 pipelined K fragments ----
    v8f s[4] = {};
    v16h k0c = load_b32x16(Kp + (size_t)kk * HD + 0, HD, lane);
    v16h k1c = load_b32x16(Kp + (size_t)kk * HD + 32, HD, lane);
#pragma unroll
    for (int j = 0; j < 4; ++j) {
      v16h k0n, k1n;
      if (j < 3) {
        k0n = load_b32x16(Kp + (size_t)(kk + (j + 1) * 16) * HD + 0, HD, lane);
        k1n = load_b32x16(Kp + (size_t)(kk + (j + 1) * 16) * HD + 32, HD, lane);
      }
      s[j] = wmma16(qa0, k0c, s[j]);
      s[j] = wmma16(qa1, k1c, s[j]);
      if (j < 3) { k0c = k0n; k1c = k1n; }
    }
    // ---- issue V loads early; softmax VALU hides their latency ----
    v16h vb0[4], vb1[4];
#pragma unroll
    for (int j = 0; j < 4; ++j) {
      vb0[j] = load_b32x16(Vp + (size_t)(j * 16) * TT + kk + 0, TT, lane);
      vb1[j] = load_b32x16(Vp + (size_t)(j * 16) * TT + kk + 32, TT, lane);
    }
    // ---- online softmax ----
#pragma unroll
    for (int r = 0; r < 8; ++r) {
      float tm = -1e30f;
#pragma unroll
      for (int j = 0; j < 4; ++j) { s[j][r] *= 0.125f; tm = fmaxf(tm, s[j][r]); }
#pragma unroll
      for (int off = 1; off < 16; off <<= 1)
        tm = fmaxf(tm, __shfl_xor(tm, off, 32));
      float nm = fmaxf(mrow[r], tm);
      float corr = __expf(mrow[r] - nm);
      mrow[r] = nm;
      float ps = 0.f;
#pragma unroll
      for (int j = 0; j < 4; ++j) {
        float p = __expf(s[j][r] - nm);
        s[j][r] = p; ps += p;
      }
      lrow[r] = lrow[r] * corr + ps;
#pragma unroll
      for (int j = 0; j < 4; ++j) o[j][r] *= corr;
    }
    // ---- C-layout -> A-layout via per-wave LDS region ----
#pragma unroll
    for (int r = 0; r < 8; ++r) {
      int mloc = r + 8 * half;
#pragma unroll
      for (int j = 0; j < 4; ++j)
        myl[mloc * HD + j * 16 + (lane & 15)] = (_Float16)s[j][r];
    }
    int am = lane & 15, akb = (lane >> 4) << 3;
    union { v16h v; v8h h2[2]; } pa0, pa1;
    pa0.h2[0] = *(const v8h*)(myl + am * HD + akb);
    pa0.h2[1] = *(const v8h*)(myl + am * HD + akb + 16);
    pa1.h2[0] = *(const v8h*)(myl + am * HD + 32 + akb);
    pa1.h2[1] = *(const v8h*)(myl + am * HD + 32 + akb + 16);
    // ---- O += P * V ----
#pragma unroll
    for (int j = 0; j < 4; ++j) {
      o[j] = wmma16(pa0.v, vb0[j], o[j]);
      o[j] = wmma16(pa1.v, vb1[j], o[j]);
    }
  }
#pragma unroll
  for (int r = 0; r < 8; ++r) {
    float l = lrow[r];
#pragma unroll
    for (int off = 1; off < 16; off <<= 1) l += __shfl_xor(l, off, 32);
    float inv = 1.f / l;
    int t = qbase + r + 8 * half;
#pragma unroll
    for (int j = 0; j < 4; ++j)
      Oa[((size_t)b * TT + t) * DD + h * HD + j * 16 + (lane & 15)] =
          (_Float16)(o[j][r] * inv);
  }
}

// ---------- output projection ----------
// grid: (MROWS/256, DD/64)  block: 256
__global__ __launch_bounds__(256) void outproj_gemm(
    const _Float16* __restrict__ A,   // [BB*TT, DD]
    const _Float16* __restrict__ W,   // [DD, DD]
    const float*    __restrict__ bias,
    float* __restrict__ out) {        // (T,B,D) fp32
  __shared__ __align__(16) _Float16 bsm[2 * 64 * 128];
  const int tid   = threadIdx.x;
  const int lane  = tid & 31;
  const int wave  = tid >> 5;
  const int mbase = blockIdx.x * 256 + wave * 32;
  const int nbase = blockIdx.y * 64;
  const _Float16* Arows = A + (size_t)mbase * DD;
  const _Float16* wbase = W + (size_t)nbase * DD;

  v8f c[2][4] = {};
  gemm_core(Arows, wbase, bsm, tid, lane, c);

  const int half = lane >> 4;
#pragma unroll
  for (int j = 0; j < 4; ++j) {
    int n = nbase + j * 16 + (lane & 15);
    float bv = bias[n];
#pragma unroll
    for (int mt = 0; mt < 2; ++mt)
#pragma unroll
      for (int r = 0; r < 8; ++r) {
        int m = mbase + mt * 16 + r + 8 * half;   // m = b*TT + t
        int b = m >> 10, t = m & 1023;
        out[((size_t)(t * BB + b)) * DD + n] = c[mt][j][r] + bv;
      }
  }
}

// ---------- host launcher ----------
extern "C" void kernel_launch(void* const* d_in, const int* in_sizes, int n_in,
                              void* d_out, int out_size, void* d_ws, size_t ws_size,
                              hipStream_t stream) {
  const float* q_in = (const float*)d_in[0];
  const float* k_in = (const float*)d_in[1];
  const float* v_in = (const float*)d_in[2];
  const float* wqkv = (const float*)d_in[3];
  const float* bqkv = (const float*)d_in[4];
  const float* ow   = (const float*)d_in[5];
  const float* ob   = (const float*)d_in[6];
  float* out = (float*)d_out;

  const size_t NX = (size_t)MROWS * DD;
  char* ws = (char*)d_ws;
  _Float16* Xh  = (_Float16*)(ws);
  _Float16* Wh  = (_Float16*)(ws + 3 * NX * 2);
  _Float16* OWh = (_Float16*)(ws + 3 * NX * 2 + 3ull * DD * DD * 2);
  char* p = ws + 3 * NX * 2 + 4ull * DD * DD * 2;
  _Float16* Qb  = (_Float16*)(p);  p += NX * 2;
  _Float16* Kb  = (_Float16*)(p);  p += NX * 2;
  _Float16* Vtb = (_Float16*)(p);  p += NX * 2;
  _Float16* Oa  = (_Float16*)(p);

  const int CB = 256;
  cvt_f32_to_f16<<<(int)((NX + CB - 1) / CB), CB, 0, stream>>>(q_in, Xh, (int)NX);
  cvt_f32_to_f16<<<(int)((NX + CB - 1) / CB), CB, 0, stream>>>(k_in, Xh + NX, (int)NX);
  cvt_f32_to_f16<<<(int)((NX + CB - 1) / CB), CB, 0, stream>>>(v_in, Xh + 2 * NX, (int)NX);
  cvt_f32_to_f16<<<(3 * DD * DD + CB - 1) / CB, CB, 0, stream>>>(wqkv, Wh, 3 * DD * DD);
  cvt_f32_to_f16<<<(DD * DD + CB - 1) / CB, CB, 0, stream>>>(ow, OWh, DD * DD);

  qkv_rope_gemm<<<dim3(MROWS / 256, DD / 64, 3), 256, 0, stream>>>(
      Xh, Wh, bqkv, Qb, Kb, Vtb);
  attn_kernel<<<dim3(TT / 128, BB * HH), 256, 0, stream>>>(Qb, Kb, Vtb, Oa);
  outproj_gemm<<<dim3(MROWS / 256, DD / 64), 256, 0, stream>>>(Oa, OWh, ob, out);
}
